// Transformer_16054587752876
// MI455X (gfx1250) — matrix operand order repaired and checked by reading, not compiled, not run on hardware
//
#include <hip/hip_runtime.h>
#include <math.h>

// ---------------------------------------------------------------------------
// Transformer encoder fwd on gfx1250 (MI455X): f16 WMMA GEMMs + flash attention
// B=8, S=1024, E=512, H=8, D=64, L=6, HF=2048, fp32 I/O, f32 accumulate.
// GEMM stages weight tiles into LDS via GLOBAL_LOAD_ASYNC_TO_LDS_B128
// (ASYNCcnt-tracked, double buffered) — the CDNA5 async data-movement path.
// ---------------------------------------------------------------------------

typedef _Float16 half_t;
typedef __attribute__((ext_vector_type(16))) _Float16 v16h;
typedef __attribute__((ext_vector_type(8)))  float    v8f;

constexpr int kB  = 8;
constexpr int kS  = 1024;
constexpr int kE  = 512;
constexpr int kH  = 8;
constexpr int kD  = 64;
constexpr int kL  = 6;
constexpr int kHF = 2048;

constexpr long kBS   = (long)kB * kS;          // 8192 rows
constexpr long kXSZ  = kBS * kE;               // activation elems
constexpr long kHSZ  = kBS * kHF;              // ffn hidden elems
constexpr long kWESZ = (long)kL * kE * kE;     // per-tensor E x E weights
constexpr long kWFSZ = (long)kL * kHF * kE;    // per-tensor HF x E weights

// ---------------------------------------------------------------------------
// CDNA5 async copy: global -> LDS, 16 bytes per lane, tracked by ASYNCcnt.
// lds_off is the 32-bit LDS byte offset (low word of a generic shared ptr).
// ---------------------------------------------------------------------------
__device__ __forceinline__ void async_ld_b128(unsigned lds_off, const half_t* g) {
  asm volatile("global_load_async_to_lds_b128 %0, %1, off"
               :: "v"(lds_off), "v"(g) : "memory");
}
__device__ __forceinline__ void wait_async0() {
  asm volatile("s_wait_asynccnt 0" ::: "memory");
}

// ---------------------------------------------------------------------------
// WMMA fragment loaders (wave32, contiguous k order k = 16*(lane/16) + e in A and B)
// ---------------------------------------------------------------------------

// A-matrix 16x32 f16: lanes 0-15 & 16-31 both hold rows M=lane%16.
// Element e: K = 16*(lane/16) + e (contiguous order, matches the B loaders).
__device__ __forceinline__ v16h load_a16x32(const half_t* __restrict__ X, int ld) {
  const int lane = threadIdx.x & 31;
  const int m  = lane & 15;
  const int hl = lane >> 4;
  v16h a;
#pragma unroll
  for (int e = 0; e < 16; ++e) {
    const int k = 16 * hl + e;
    a[e] = X[(long)m * ld + k];
  }
  return a;
}

// B-matrix 32x16 f16 from a row-major [N x K] tile, transposed on the fly:
// B[k][n] = W[n][k]. lanes 0-15: K=0..15, lanes 16-31: K=16..31, N = lane%16.
__device__ __forceinline__ v16h load_bT32x16(const half_t* __restrict__ W, int ld) {
  const int lane = threadIdx.x & 31;
  const int n  = lane & 15;
  const int kb = (lane >> 4) << 4;
  v16h b;
#pragma unroll
  for (int e = 0; e < 16; ++e) b[e] = W[(long)n * ld + kb + e];
  return b;
}

// B-matrix 32x16 f16 from a row-major [K x N] tile (no transpose): B[k][n] = V[k][n].
__device__ __forceinline__ v16h load_b32x16(const half_t* __restrict__ V, int ld) {
  const int lane = threadIdx.x & 31;
  const int n  = lane & 15;
  const int kb = (lane >> 4) << 4;
  v16h b;
#pragma unroll
  for (int e = 0; e < 16; ++e) b[e] = V[(long)(kb + e) * ld + n];
  return b;
}

__device__ __forceinline__ float gelu_tanh(float v) {
  const float c0 = 0.7978845608028654f; // sqrt(2/pi)
  const float u = c0 * (v + 0.044715f * v * v * v);
  return 0.5f * v * (1.0f + tanhf(u));
}

// ---------------------------------------------------------------------------
// GEMM: Y[M,N] = act(X[M,K] * W[N,K]^T + bias[N]).
// Block = 4 waves = 128x64 output tile; each wave owns 32x64 (8 WMMAs/K-step).
// The 64x32 B tile is staged into LDS with async-to-LDS, double buffered:
// LDS layout Bs[buf][col*32 + k] so a B fragment is 2 contiguous ds_load_b128.
// ---------------------------------------------------------------------------
__global__ __launch_bounds__(128) void gemm_xwt(
    const half_t* __restrict__ X, const half_t* __restrict__ W,
    const float* __restrict__ bias, half_t* __restrict__ Yh,
    float* __restrict__ Yf, int M, int N, int K, int act)
{
  __shared__ half_t Bs[2][64 * 32];

  const int wave = threadIdx.x >> 5;
  const long row0 = (long)blockIdx.y * 128 + wave * 32;
  const long col0 = (long)blockIdx.x * 64;

  // async staging assignment: thread -> (column, 16-wide k segment)
  const int scol = threadIdx.x >> 1;
  const int sk16 = (threadIdx.x & 1) << 4;

  v8f acc[2][4];
#pragma unroll
  for (int s2 = 0; s2 < 2; ++s2)
#pragma unroll
    for (int t = 0; t < 4; ++t)
#pragma unroll
      for (int r = 0; r < 8; ++r) acc[s2][t][r] = 0.0f;

  // prefetch tile 0 into buffer 0
  {
    const half_t* gsrc = W + (col0 + scol) * (long)K + sk16;
    const unsigned l0 =
        (unsigned)(unsigned long long)&Bs[0][scol * 32 + sk16];
    async_ld_b128(l0, gsrc);
    async_ld_b128(l0 + 16, gsrc + 8);
  }

  int buf = 0;
  for (int k0 = 0; k0 < K; k0 += 32) {
    wait_async0();
    __syncthreads();

    // prefetch next tile into the other buffer (all waves already consumed it)
    if (k0 + 32 < K) {
      const half_t* gsrc = W + (col0 + scol) * (long)K + (k0 + 32) + sk16;
      const unsigned l0 =
          (unsigned)(unsigned long long)&Bs[buf ^ 1][scol * 32 + sk16];
      async_ld_b128(l0, gsrc);
      async_ld_b128(l0 + 16, gsrc + 8);
    }

    const v16h a0 = load_a16x32(X + row0 * K + k0, K);
    const v16h a1 = load_a16x32(X + (row0 + 16) * K + k0, K);
#pragma unroll
    for (int t = 0; t < 4; ++t) {
      // B fragment from LDS: Bt[n*32 + k], 16 contiguous halves per lane
      const half_t* Bt = &Bs[buf][t * 16 * 32];
      const int lane = threadIdx.x & 31;
      const int n = lane & 15;
      const int kb = (lane >> 4) << 4;
      v16h b;
#pragma unroll
      for (int e = 0; e < 16; ++e) b[e] = Bt[n * 32 + kb + e];
      acc[0][t] = __builtin_amdgcn_wmma_f32_16x16x32_f16(
          false, a0, false, b, (short)0, acc[0][t], false, false);
      acc[1][t] = __builtin_amdgcn_wmma_f32_16x16x32_f16(
          false, a1, false, b, (short)0, acc[1][t], false, false);
    }
    buf ^= 1;
  }

  // C/D layout: VGPR r -> M = r + 8*(lane/16), N = lane%16.
  const int lane = threadIdx.x & 31;
  const int n = lane & 15, hl = lane >> 4;
#pragma unroll
  for (int s2 = 0; s2 < 2; ++s2) {
#pragma unroll
    for (int t = 0; t < 4; ++t) {
#pragma unroll
      for (int r = 0; r < 8; ++r) {
        const long row = row0 + s2 * 16 + 8 * hl + r;
        const long col = col0 + t * 16 + n;
        float v = acc[s2][t][r] + bias[col];
        if (act) v = gelu_tanh(v);
        if (Yh) Yh[row * N + col] = (half_t)v;
        if (Yf) Yf[row * N + col] = v;
      }
    }
  }
}

// ---------------------------------------------------------------------------
// Flash attention: 1 wave per (b, h, 16 query rows). Streams over keys in
// blocks of 32; online softmax; P goes through LDS to transpose C-layout ->
// A-fragment layout for the P*V WMMAs. Mask: score - (!mask)*1e6, scale 1/8.
// ---------------------------------------------------------------------------
__global__ __launch_bounds__(32) void flash_attn(
    const half_t* __restrict__ Q, const half_t* __restrict__ Kc,
    const half_t* __restrict__ Vc, const int* __restrict__ mask,
    half_t* __restrict__ O)
{
  __shared__ half_t P[16 * 32];

  const int b = blockIdx.z, h = blockIdx.y;
  const long q0 = (long)blockIdx.x * 16;
  const int lane = threadIdx.x & 31;
  const int n = lane & 15, hl = lane >> 4;

  const half_t* qp = Q + ((long)b * kS + q0) * kE + h * kD;
  const v16h aq0 = load_a16x32(qp, kE);        // d = 0..31
  const v16h aq1 = load_a16x32(qp + 32, kE);   // d = 32..63

  float mrow[8], lrow[8];
  v8f o[4];
#pragma unroll
  for (int r = 0; r < 8; ++r) { mrow[r] = -1e30f; lrow[r] = 0.0f; }
#pragma unroll
  for (int t = 0; t < 4; ++t)
#pragma unroll
    for (int r = 0; r < 8; ++r) o[t][r] = 0.0f;

  for (int kb = 0; kb < kS; kb += 32) {
    float s[2][8];
#pragma unroll
    for (int sub = 0; sub < 2; ++sub) {
      const half_t* kp = Kc + ((long)b * kS + kb + sub * 16) * kE + h * kD;
      const v16h bk0 = load_bT32x16(kp, kE);       // B[d][key] over d=0..31
      const v16h bk1 = load_bT32x16(kp + 32, kE);  // d=32..63
      v8f z;
#pragma unroll
      for (int r = 0; r < 8; ++r) z[r] = 0.0f;
      z = __builtin_amdgcn_wmma_f32_16x16x32_f16(false, aq0, false, bk0, (short)0, z, false, false);
      z = __builtin_amdgcn_wmma_f32_16x16x32_f16(false, aq1, false, bk1, (short)0, z, false, false);
      const float mpen =
          (mask[(long)b * kS + kb + sub * 16 + n] != 0) ? 0.0f : 1.0e6f;
#pragma unroll
      for (int r = 0; r < 8; ++r) s[sub][r] = z[r] * 0.125f - mpen;  // 1/sqrt(64)
    }

    // Online softmax: reduce over the 16 key columns (lanes within half-group).
#pragma unroll
    for (int r = 0; r < 8; ++r) {
      float mx = fmaxf(s[0][r], s[1][r]);
#pragma unroll
      for (int msk = 1; msk < 16; msk <<= 1) mx = fmaxf(mx, __shfl_xor(mx, msk, 32));
      const float mnew = fmaxf(mrow[r], mx);
      const float corr = __expf(mrow[r] - mnew);
      mrow[r] = mnew;
      const float p0 = __expf(s[0][r] - mnew);
      const float p1 = __expf(s[1][r] - mnew);
      float ps = p0 + p1;
#pragma unroll
      for (int msk = 1; msk < 16; msk <<= 1) ps += __shfl_xor(ps, msk, 32);
      lrow[r] = lrow[r] * corr + ps;
#pragma unroll
      for (int t = 0; t < 4; ++t) o[t][r] *= corr;
      // C-layout -> LDS (row m = r + 8*hl, 32 key columns)
      P[(r + 8 * hl) * 32 + n]      = (half_t)p0;
      P[(r + 8 * hl) * 32 + 16 + n] = (half_t)p1;
    }
    __syncthreads();

    const v16h ap = load_a16x32(P, 32);  // P as A-fragment (16 x 32 keys)
    const half_t* vp = Vc + ((long)b * kS + kb) * kE + h * kD;
#pragma unroll
    for (int t = 0; t < 4; ++t) {
      const v16h bv = load_b32x16(vp + t * 16, kE);  // 32 keys x 16 dims
      o[t] = __builtin_amdgcn_wmma_f32_16x16x32_f16(
          false, ap, false, bv, (short)0, o[t], false, false);
    }
    __syncthreads();
  }

  // Normalize and scatter back to (B, S, E) with head offset.
#pragma unroll
  for (int t = 0; t < 4; ++t)
#pragma unroll
    for (int r = 0; r < 8; ++r) {
      const long row = q0 + 8 * hl + r;
      const long col = (long)h * kD + t * 16 + n;
      O[((long)b * kS + row) * kE + col] = (half_t)(o[t][r] / lrow[r]);
    }
}

// ---------------------------------------------------------------------------
// Residual add + LayerNorm over E=512; writes f32 stream (d_out) and f16 copy.
// One 256-thread block per row, 2 elements per thread.
// ---------------------------------------------------------------------------
__global__ __launch_bounds__(256) void add_ln(
    float* __restrict__ X, const float* __restrict__ Dl,
    const float* __restrict__ g, const float* __restrict__ bt,
    half_t* __restrict__ Xh)
{
  __shared__ float red[256];
  const long row = blockIdx.x;
  const int t = threadIdx.x;

  const float v0 = X[row * kE + t]       + Dl[row * kE + t];
  const float v1 = X[row * kE + t + 256] + Dl[row * kE + t + 256];

  red[t] = v0 + v1;
  __syncthreads();
  for (int s2 = 128; s2 > 0; s2 >>= 1) {
    if (t < s2) red[t] += red[t + s2];
    __syncthreads();
  }
  const float mean = red[0] / (float)kE;
  __syncthreads();

  const float d0 = v0 - mean, d1 = v1 - mean;
  red[t] = d0 * d0 + d1 * d1;
  __syncthreads();
  for (int s2 = 128; s2 > 0; s2 >>= 1) {
    if (t < s2) red[t] += red[t + s2];
    __syncthreads();
  }
  const float rstd = rsqrtf(red[0] / (float)kE + 1e-5f);

  const float y0 = g[t] * d0 * rstd + bt[t];
  const float y1 = g[t + 256] * d1 * rstd + bt[t + 256];
  X[row * kE + t]        = y0;
  X[row * kE + t + 256]  = y1;
  Xh[row * kE + t]       = (half_t)y0;
  Xh[row * kE + t + 256] = (half_t)y1;
}

// ---------------------------------------------------------------------------
// Elementwise helpers
// ---------------------------------------------------------------------------
__global__ void cvt_h(const float* __restrict__ s, half_t* __restrict__ d, long n) {
  long i = (long)blockIdx.x * blockDim.x + threadIdx.x;
  const long st = (long)gridDim.x * blockDim.x;
  for (; i < n; i += st) d[i] = (half_t)s[i];
}

__global__ void init_x(const float* __restrict__ xin, float* __restrict__ x,
                       half_t* __restrict__ xh, long n) {
  long i = (long)blockIdx.x * blockDim.x + threadIdx.x;
  const long st = (long)gridDim.x * blockDim.x;
  for (; i < n; i += st) {
    const float v = xin[i];
    x[i] = v;
    xh[i] = (half_t)v;
  }
}

// ---------------------------------------------------------------------------
// Host driver
// ---------------------------------------------------------------------------
extern "C" void kernel_launch(void* const* d_in, const int* in_sizes, int n_in,
                              void* d_out, int out_size, void* d_ws, size_t ws_size,
                              hipStream_t stream) {
  (void)in_sizes; (void)n_in; (void)out_size; (void)ws_size;

  const float* x_in = (const float*)d_in[0];
  const int*   att  = (const int*)  d_in[1];
  const float* WQ = (const float*)d_in[2];  const float* bQ = (const float*)d_in[3];
  const float* WK = (const float*)d_in[4];  const float* bK = (const float*)d_in[5];
  const float* WV = (const float*)d_in[6];  const float* bV = (const float*)d_in[7];
  const float* WO = (const float*)d_in[8];  const float* bO = (const float*)d_in[9];
  const float* W1 = (const float*)d_in[10]; const float* b1 = (const float*)d_in[11];
  const float* W2 = (const float*)d_in[12]; const float* b2 = (const float*)d_in[13];
  const float* g1 = (const float*)d_in[14]; const float* bt1 = (const float*)d_in[15];
  const float* g2 = (const float*)d_in[16]; const float* bt2 = (const float*)d_in[17];

  float* X = (float*)d_out;  // f32 residual stream lives in d_out

  // Workspace carve-up (f16 buffers first, f32 scratch after)
  half_t* w = (half_t*)d_ws;
  half_t* xh  = w; w += kXSZ;
  half_t* qh  = w; w += kXSZ;
  half_t* kh  = w; w += kXSZ;
  half_t* vh  = w; w += kXSZ;
  half_t* ah  = w; w += kXSZ;
  half_t* ffh = w; w += kHSZ;
  half_t* WQh = w; w += kWESZ;
  half_t* WKh = w; w += kWESZ;
  half_t* WVh = w; w += kWESZ;
  half_t* WOh = w; w += kWESZ;
  half_t* W1h = w; w += kWFSZ;
  half_t* W2h = w; w += kWFSZ;
  float*  proj = (float*)w;  // kXSZ floats

  // Weight conversion (once; graph replays keep it deterministic)
  cvt_h<<<2048, 256, 0, stream>>>(WQ, WQh, kWESZ);
  cvt_h<<<2048, 256, 0, stream>>>(WK, WKh, kWESZ);
  cvt_h<<<2048, 256, 0, stream>>>(WV, WVh, kWESZ);
  cvt_h<<<2048, 256, 0, stream>>>(WO, WOh, kWESZ);
  cvt_h<<<2048, 256, 0, stream>>>(W1, W1h, kWFSZ);
  cvt_h<<<2048, 256, 0, stream>>>(W2, W2h, kWFSZ);
  init_x<<<2048, 256, 0, stream>>>(x_in, X, xh, kXSZ);

  const dim3 gE(kE / 64,  (int)(kBS / 128));  // 8  x 64 blocks
  const dim3 gF(kHF / 64, (int)(kBS / 128));  // 32 x 64 blocks
  const dim3 gA(kS / 16, kH, kB);             // flash attention grid

  for (int l = 0; l < kL; ++l) {
    const long wE = (long)l * kE * kE;
    const long wF = (long)l * kHF * kE;

    // Q, K, V projections (f16 out)
    gemm_xwt<<<gE, 128, 0, stream>>>(xh, WQh + wE, bQ + (long)l * kE, qh, nullptr,
                                     (int)kBS, kE, kE, 0);
    gemm_xwt<<<gE, 128, 0, stream>>>(xh, WKh + wE, bK + (long)l * kE, kh, nullptr,
                                     (int)kBS, kE, kE, 0);
    gemm_xwt<<<gE, 128, 0, stream>>>(xh, WVh + wE, bV + (long)l * kE, vh, nullptr,
                                     (int)kBS, kE, kE, 0);

    // Multi-head attention (flash, no S x S materialization)
    flash_attn<<<gA, 32, 0, stream>>>(qh, kh, vh, att, ah);

    // Output projection (f32 out for residual)
    gemm_xwt<<<gE, 128, 0, stream>>>(ah, WOh + wE, bO + (long)l * kE, nullptr, proj,
                                     (int)kBS, kE, kE, 0);
    add_ln<<<(int)kBS, 256, 0, stream>>>(X, proj, g1 + (long)l * kE,
                                         bt1 + (long)l * kE, xh);

    // FFN: GELU(x W1^T + b1) W2^T + b2
    gemm_xwt<<<gF, 128, 0, stream>>>(xh, W1h + wF, b1 + (long)l * kHF, ffh, nullptr,
                                     (int)kBS, kHF, kE, 1);
    gemm_xwt<<<gE, 128, 0, stream>>>(ffh, W2h + wF, b2 + (long)l * kE, nullptr, proj,
                                     (int)kBS, kE, kHF, 0);
    add_ln<<<(int)kBS, 256, 0, stream>>>(X, proj, g2 + (long)l * kE,
                                         bt2 + (long)l * kE, xh);
  }
}
